// HierarchicalMemoryTransformer_47081431498895
// MI455X (gfx1250) — matrix operand order repaired and checked by reading, not compiled, not run on hardware
//
#include <hip/hip_runtime.h>

// ---------------------------------------------------------------------------
// HierarchicalMemoryTransformer for MI455X (gfx1250), wave32 + WMMA f16.
//   1. cvt4:      fp32 -> f16 for Wq/Wk/Wv/Wo and retrieved_memory
//   2. ln_kernel: LayerNorm(h) -> h16 (f16)
//   3. gemm_nt:   q/k/v projections (WMMA, async-LDS 3-buffer pipeline)
//   4. vtrans:    v16 -> vT16 [B,H,D,K]
//   5. attn_kernel: scores WMMA -> softmax -> attn@V WMMA -> o16 (f16)
//   6. gemm_nt mode1: d_out = h + sigmoid(gate) * (o16 @ Wo^T + bo)  (f32)
// ---------------------------------------------------------------------------

typedef __attribute__((ext_vector_type(16))) _Float16 v16h;
typedef __attribute__((ext_vector_type(8)))  _Float16 v8h;
typedef __attribute__((ext_vector_type(4)))  _Float16 v4h;
typedef __attribute__((ext_vector_type(8)))  float    v8f;
typedef __attribute__((ext_vector_type(4)))  int      v4i;

// ---- CDNA5 async global->LDS copy (guarded; fallback = sync staging) ------
#ifndef HAVE_ASYNC_LDS
#  if defined(__has_builtin)
#    if __has_builtin(__builtin_amdgcn_global_load_async_to_lds_b128)
#      define HAVE_ASYNC_LDS 1
#    else
#      define HAVE_ASYNC_LDS 0
#    endif
#  else
#    define HAVE_ASYNC_LDS 0
#  endif
#endif

#if HAVE_ASYNC_LDS
typedef __attribute__((address_space(1))) v4i g_v4i;   // global 128-bit chunk
typedef __attribute__((address_space(3))) v4i l_v4i;   // LDS 128-bit chunk

__device__ __forceinline__ void async_cp16(const _Float16* g, _Float16* l) {
  __builtin_amdgcn_global_load_async_to_lds_b128(
      (g_v4i*)(void*)const_cast<_Float16*>(g), (l_v4i*)(void*)l, 0, 0);
}

template <int N>
__device__ __forceinline__ void wait_async() {
#if defined(__has_builtin)
#if __has_builtin(__builtin_amdgcn_s_wait_asynccnt)
  __builtin_amdgcn_s_wait_asynccnt(N);
#else
  asm volatile("s_wait_asynccnt %0" ::"i"(N) : "memory");
#endif
#else
  asm volatile("s_wait_asynccnt %0" ::"i"(N) : "memory");
#endif
}
#endif  // HAVE_ASYNC_LDS

__device__ __forceinline__ v8f wmma_f16(v16h a, v16h b, v8f c) {
  return __builtin_amdgcn_wmma_f32_16x16x32_f16(
      /*neg_a=*/false, a, /*neg_b=*/false, b,
      /*c_mod=*/(short)0, c, /*reuse_a=*/false, /*reuse_b=*/false);
}

// Load a 16-half WMMA fragment as two 8-half chunks (lo at p, hi at p+hioff).
// A-frag and B-frag: hioff = 16 (K and K+16 chunks).
__device__ __forceinline__ v16h ldfrag(const _Float16* p, int hioff) {
  v8h lo = *(const v8h*)p;
  v8h hi = *(const v8h*)(p + hioff);
  v16h f;
#pragma unroll
  for (int e = 0; e < 8; ++e) { f[e] = lo[e]; f[e + 8] = hi[e]; }
  return f;
}

// ---------------------------------------------------------------------------
// fp32 -> f16 convert, 4 elements/thread
// ---------------------------------------------------------------------------
__global__ __launch_bounds__(256) void cvt4(const float* __restrict__ in,
                                            _Float16* __restrict__ out, int n4) {
  int i = blockIdx.x * 256 + threadIdx.x;
  if (i < n4) {
    float4 f = ((const float4*)in)[i];
    v4h h = { (_Float16)f.x, (_Float16)f.y, (_Float16)f.z, (_Float16)f.w };
    *(v4h*)(out + 4ull * i) = h;
  }
}

// ---------------------------------------------------------------------------
// LayerNorm over d=1024, one row per block (256 threads, 4 elems/thread)
// ---------------------------------------------------------------------------
__global__ __launch_bounds__(256) void ln_kernel(const float* __restrict__ x,
                                                 const float* __restrict__ g,
                                                 const float* __restrict__ bb,
                                                 _Float16* __restrict__ o) {
  constexpr int D = 1024;
  int row = blockIdx.x, tid = threadIdx.x;
  const float* xr = x + (size_t)row * D;
  float v[4];
  float s = 0.f;
#pragma unroll
  for (int i = 0; i < 4; ++i) { v[i] = xr[tid + i * 256]; s += v[i]; }
  __shared__ float red[8];
#pragma unroll
  for (int m = 16; m >= 1; m >>= 1) s += __shfl_xor(s, m, 32);
  if (!(tid & 31)) red[tid >> 5] = s;
  __syncthreads();
  s = 0.f;
#pragma unroll
  for (int i = 0; i < 8; ++i) s += red[i];
  float mu = s * (1.f / 1024.f);
  __syncthreads();
  float q = 0.f;
#pragma unroll
  for (int i = 0; i < 4; ++i) { float d = v[i] - mu; q += d * d; }
#pragma unroll
  for (int m = 16; m >= 1; m >>= 1) q += __shfl_xor(q, m, 32);
  if (!(tid & 31)) red[tid >> 5] = q;
  __syncthreads();
  q = 0.f;
#pragma unroll
  for (int i = 0; i < 8; ++i) q += red[i];
  float rs = rsqrtf(q * (1.f / 1024.f) + 1e-5f);
#pragma unroll
  for (int i = 0; i < 4; ++i) {
    int c = tid + i * 256;
    o[(size_t)row * D + c] = (_Float16)((v[i] - mu) * rs * g[c] + bb[c]);
  }
}

// ---------------------------------------------------------------------------
// NT GEMM: C[M,N] = A[M,K] * B[N,K]^T, K = 1024, f16 inputs, f32 accum.
// Block: 256 threads (8 waves), 128x128 C tile; wave = 64x32 (4x2 WMMA tiles).
// Async path: 3 LDS buffers, prefetch distance 1, one barrier per k-step.
// mode 0: out16[idx] = f16(acc + bias[n])   (LDS-repacked b128 stores)
// mode 1: outf[idx]  = resid[idx] + gate * (acc + bias[n])
// ---------------------------------------------------------------------------
#define LDK 40            // padded LDS row stride in halves (80B)
#define TILEH (128 * LDK) // halves per 128x32 staged tile = 5120

__global__ __launch_bounds__(256) void gemm_nt(
    const _Float16* __restrict__ A, const _Float16* __restrict__ B,
    const float* __restrict__ bias, _Float16* __restrict__ out16,
    float* __restrict__ outf, const float* __restrict__ resid,
    const float* __restrict__ gatep, int mode) {
  constexpr int K = 1024;
  constexpr int N = 1024;
  constexpr int NK = K / 32;

  int tid = threadIdx.x;
  int lane = tid & 31, w = tid >> 5;
  int m0 = blockIdx.x * 128, n0 = blockIdx.y * 128;
  int wm = (w >> 2) * 64, wn = (w & 3) * 32;

  v8f acc[4][2];
#pragma unroll
  for (int i = 0; i < 4; ++i)
#pragma unroll
    for (int j = 0; j < 2; ++j) acc[i][j] = (v8f)0.f;

  int lr = tid >> 2;          // staging row 0..63
  int seg = (tid & 3) * 8;    // staging 8-half segment
  int frow = lane & 15;
  int fko = (lane >> 4) * 8;  // A-frag K sub-offset per half-wave
  int bko = (lane >> 4) * 8;  // B-frag K sub-offset per half-wave

#if HAVE_ASYNC_LDS
  // Unified LDS: 3 A buffers + 3 B buffers; epilogue repack aliases it.
  __shared__ __align__(16) _Float16 sM[6 * TILEH];  // 61440 B
#define SA(buf) (&sM[(buf) * TILEH])
#define SB(buf) (&sM[(3 + (buf)) * TILEH])

  // Prologue: async-stage tile 0 into buffer 0 (4 x b128 per thread).
#pragma unroll
  for (int it = 0; it < 2; ++it) {
    int row = lr + it * 64;
    async_cp16(&A[(size_t)(m0 + row) * K + seg], SA(0) + row * LDK + seg);
    async_cp16(&B[(size_t)(n0 + row) * K + seg], SB(0) + row * LDK + seg);
  }

  for (int k = 0; k < NK; ++k) {
    int cur = k % 3;
    if (k + 1 < NK) {
      int nb = (k + 1) % 3;
      int k0 = (k + 1) * 32;
#pragma unroll
      for (int it = 0; it < 2; ++it) {
        int row = lr + it * 64;
        async_cp16(&A[(size_t)(m0 + row) * K + k0 + seg], SA(nb) + row * LDK + seg);
        async_cp16(&B[(size_t)(n0 + row) * K + k0 + seg], SB(nb) + row * LDK + seg);
      }
      wait_async<4>();   // tile k resident; tile k+1 still in flight
    } else {
      wait_async<0>();
    }
    __syncthreads();     // tile k visible from every wave's copies

    v16h af[4], bf[2];
#pragma unroll
    for (int i = 0; i < 4; ++i)
      af[i] = ldfrag(SA(cur) + (wm + i * 16 + frow) * LDK + fko, 16);
#pragma unroll
    for (int j = 0; j < 2; ++j)
      bf[j] = ldfrag(SB(cur) + (wn + j * 16 + frow) * LDK + bko, 16);
#pragma unroll
    for (int i = 0; i < 4; ++i)
#pragma unroll
      for (int j = 0; j < 2; ++j) acc[i][j] = wmma_f16(af[i], bf[j], acc[i][j]);
  }
#else  // ---- synchronous fallback ----
  __shared__ __align__(16) _Float16 sM[4 * TILEH];  // staging(2) + repack room
#define SA(buf) (&sM[0])
#define SB(buf) (&sM[TILEH])
  for (int k0 = 0; k0 < K; k0 += 32) {
#pragma unroll
    for (int it = 0; it < 2; ++it) {
      int row = lr + it * 64;
      *(v8h*)(SA(0) + row * LDK + seg) = *(const v8h*)&A[(size_t)(m0 + row) * K + k0 + seg];
      *(v8h*)(SB(0) + row * LDK + seg) = *(const v8h*)&B[(size_t)(n0 + row) * K + k0 + seg];
    }
    __syncthreads();
    v16h af[4], bf[2];
#pragma unroll
    for (int i = 0; i < 4; ++i)
      af[i] = ldfrag(SA(0) + (wm + i * 16 + frow) * LDK + fko, 16);
#pragma unroll
    for (int j = 0; j < 2; ++j)
      bf[j] = ldfrag(SB(0) + (wn + j * 16 + frow) * LDK + bko, 16);
#pragma unroll
    for (int i = 0; i < 4; ++i)
#pragma unroll
      for (int j = 0; j < 2; ++j) acc[i][j] = wmma_f16(af[i], bf[j], acc[i][j]);
    __syncthreads();
  }
#endif

  // Epilogue. C layout: VGPR e -> (M = e (+8 if lane>=16), N = lane&15).
  int cn = lane & 15;
  int moff = (lane >> 4) * 8;
  if (mode == 0) {
    // Repack the wave's 64x32 f16 tile via LDS, then coalesced b128 stores.
    __syncthreads();  // all waves done reading staged tiles
    _Float16* sW = &sM[w * (64 * LDK)];
#pragma unroll
    for (int i = 0; i < 4; ++i) {
#pragma unroll
      for (int j = 0; j < 2; ++j) {
        float bsv = bias[n0 + wn + j * 16 + cn];
#pragma unroll
        for (int e = 0; e < 8; ++e) {
          int rloc = i * 16 + moff + e;  // 0..63
          sW[rloc * LDK + j * 16 + cn] = (_Float16)(acc[i][j][e] + bsv);
        }
      }
    }
    // Same-wave LDS in-order: read back coalesced (8 x b128 per lane).
#pragma unroll
    for (int r8 = 0; r8 < 8; ++r8) {
      int row = r8 * 8 + (lane >> 2);
      int sg = (lane & 3) * 8;
      size_t idx = (size_t)(m0 + wm + row) * N + (n0 + wn + sg);
      *(v8h*)&out16[idx] = *(const v8h*)&sW[row * LDK + sg];
    }
  } else {
    float gl = gatep[0];
    float gate = 1.f / (1.f + __expf(-gl));
#pragma unroll
    for (int i = 0; i < 4; ++i) {
#pragma unroll
      for (int j = 0; j < 2; ++j) {
        int nn = n0 + wn + j * 16 + cn;
        float bsv = bias[nn];
#pragma unroll
        for (int e = 0; e < 8; ++e) {
          int mm = m0 + wm + i * 16 + moff + e;
          size_t idx = (size_t)mm * N + nn;
          outf[idx] = resid[idx] + gate * (acc[i][j][e] + bsv);
        }
      }
    }
  }
#undef SA
#undef SB
}

// ---------------------------------------------------------------------------
// v16 [B*K, H*D] -> vT16 [B, H, D, K]   (B=8, K=64, H=16, D=64)
// ---------------------------------------------------------------------------
__global__ __launch_bounds__(256) void vtrans(const _Float16* __restrict__ v,
                                              _Float16* __restrict__ vT) {
  int idx = blockIdx.x * 256 + threadIdx.x;  // enumerates (b,kk,hh,dd)
  if (idx >= 8 * 64 * 16 * 64) return;
  int dd = idx & 63;
  int hh = (idx >> 6) & 15;
  int kk = (idx >> 10) & 63;
  int b  = idx >> 16;
  vT[(size_t)(((b * 16 + hh) * 64 + dd)) * 64 + kk] = v[idx];
}

// ---------------------------------------------------------------------------
// Attention: per (b, h), 64 rows of T per block (4 waves x 16 rows).
// scores = q·k / 8 -> softmax over 64 keys -> out = attn @ v
// ---------------------------------------------------------------------------
__global__ __launch_bounds__(128) void attn_kernel(
    const _Float16* __restrict__ q16, const _Float16* __restrict__ k16,
    const _Float16* __restrict__ vT16, _Float16* __restrict__ o16) {
  __shared__ __align__(16) _Float16 sP[4][16][72];
  int tid = threadIdx.x;
  int w = tid >> 5, lane = tid & 31;
  int bid = blockIdx.x;
  int tt = bid & 31;
  int hh = (bid >> 5) & 15;
  int b  = bid >> 9;
  int t0 = tt * 64 + w * 16;

  const _Float16* qp = q16 + ((size_t)(b * 2048 + t0)) * 1024 + hh * 64;
  const _Float16* kp = k16 + ((size_t)(b * 64)) * 1024 + hh * 64;
  const _Float16* vp = vT16 + (size_t)((b * 16 + hh) * 64) * 64;

  int frow = lane & 15;
  int fko = (lane >> 4) * 8;
  int bko = (lane >> 4) * 8;
  int crow = (lane >> 4) * 8;  // C-layout row offset for this half-wave

  // ---- scores: 16 T-rows x 64 keys, accumulate over D=64 (2 k-steps) ----
  v8f sc[4];
#pragma unroll
  for (int nt = 0; nt < 4; ++nt) sc[nt] = (v8f)0.f;
#pragma unroll
  for (int ks = 0; ks < 2; ++ks) {
    int db = ks * 32;
    v16h af = ldfrag(qp + (size_t)frow * 1024 + db + fko, 16);
#pragma unroll
    for (int nt = 0; nt < 4; ++nt) {
      v16h bf = ldfrag(kp + (size_t)(nt * 16 + frow) * 1024 + db + bko, 16);
      sc[nt] = wmma_f16(af, bf, sc[nt]);
    }
  }

  // ---- softmax over 64 keys (4 tiles x 16 lanes of a half-wave) ----
  v8f p[4];
#pragma unroll
  for (int e = 0; e < 8; ++e) {
    float mx = -3.0e38f;
#pragma unroll
    for (int nt = 0; nt < 4; ++nt) mx = fmaxf(mx, sc[nt][e] * 0.125f);
#pragma unroll
    for (int msk = 8; msk >= 1; msk >>= 1) mx = fmaxf(mx, __shfl_xor(mx, msk, 32));
    float sum = 0.f;
#pragma unroll
    for (int nt = 0; nt < 4; ++nt) {
      float pe = __expf(sc[nt][e] * 0.125f - mx);
      p[nt][e] = pe;
      sum += pe;
    }
#pragma unroll
    for (int msk = 8; msk >= 1; msk >>= 1) sum += __shfl_xor(sum, msk, 32);
    float r = 1.f / sum;
#pragma unroll
    for (int nt = 0; nt < 4; ++nt) p[nt][e] *= r;
  }

  // ---- C-layout -> A-layout via LDS ----
#pragma unroll
  for (int nt = 0; nt < 4; ++nt)
#pragma unroll
    for (int e = 0; e < 8; ++e)
      sP[w][crow + e][nt * 16 + frow] = (_Float16)p[nt][e];
  __syncthreads();

  // ---- out = attn(16x64) @ V(64x64), 2 k-steps over keys ----
  v8f oc[4];
#pragma unroll
  for (int nt = 0; nt < 4; ++nt) oc[nt] = (v8f)0.f;
#pragma unroll
  for (int ks = 0; ks < 2; ++ks) {
    int kb = ks * 32;
    v16h af = ldfrag(&sP[w][frow][kb + fko], 16);
#pragma unroll
    for (int nt = 0; nt < 4; ++nt) {
      v16h bf = ldfrag(vp + (size_t)(nt * 16 + frow) * 64 + kb + bko, 16);
      oc[nt] = wmma_f16(af, bf, oc[nt]);
    }
  }
  __syncthreads();

  // ---- repack to LDS, then coalesced f16 stores ----
#pragma unroll
  for (int nt = 0; nt < 4; ++nt)
#pragma unroll
    for (int e = 0; e < 8; ++e)
      sP[w][crow + e][nt * 16 + frow] = (_Float16)oc[nt][e];
  __syncthreads();

  _Float16* op = o16 + ((size_t)(b * 2048 + t0)) * 1024 + hh * 64;
#pragma unroll
  for (int it = 0; it < 4; ++it) {
    int row = it * 4 + (lane >> 3);
    int sg = (lane & 7) * 8;
    *(v8h*)&op[(size_t)row * 1024 + sg] = *(const v8h*)&sP[w][row][sg];
  }
}

// ---------------------------------------------------------------------------
// Host launcher
// ---------------------------------------------------------------------------
extern "C" void kernel_launch(void* const* d_in, const int* in_sizes, int n_in,
                              void* d_out, int out_size, void* d_ws, size_t ws_size,
                              hipStream_t stream) {
  const float* h    = (const float*)d_in[0];
  const float* mem  = (const float*)d_in[1];
  const float* Wq   = (const float*)d_in[2];
  const float* bq   = (const float*)d_in[3];
  const float* Wk   = (const float*)d_in[4];
  const float* bk   = (const float*)d_in[5];
  const float* Wv   = (const float*)d_in[6];
  const float* bv   = (const float*)d_in[7];
  const float* Wo   = (const float*)d_in[8];
  const float* bo   = (const float*)d_in[9];
  const float* lng  = (const float*)d_in[10];
  const float* lnb  = (const float*)d_in[11];
  const float* gate = (const float*)d_in[12];
  float* out = (float*)d_out;

  // Workspace layout (bytes). h16 region reused as attention output.
  char* ws = (char*)d_ws;
  _Float16* h16   = (_Float16*)(ws);                           // 32 MB (16M halves)
  _Float16* o16   = h16;                                       // reuse after q GEMM
  _Float16* q16   = (_Float16*)(ws + (32ull << 20));           // 32 MB
  _Float16* wq16  = (_Float16*)(ws + (64ull << 20));           // 2 MB each
  _Float16* wk16  = wq16 + 1024 * 1024;
  _Float16* wv16  = wk16 + 1024 * 1024;
  _Float16* wo16  = wv16 + 1024 * 1024;
  _Float16* mem16 = wo16 + 1024 * 1024;                        // 1 MB
  _Float16* k16   = mem16 + 512 * 1024;                        // 1 MB
  _Float16* v16   = k16 + 512 * 1024;                          // 1 MB
  _Float16* vT16  = v16 + 512 * 1024;                          // 1 MB

  // 1) convert weights + memory to f16
  cvt4<<<1024, 256, 0, stream>>>(Wq, wq16, 262144);
  cvt4<<<1024, 256, 0, stream>>>(Wk, wk16, 262144);
  cvt4<<<1024, 256, 0, stream>>>(Wv, wv16, 262144);
  cvt4<<<1024, 256, 0, stream>>>(Wo, wo16, 262144);
  cvt4<<<512, 256, 0, stream>>>(mem, mem16, 131072);

  // 2) LayerNorm(h) -> h16
  ln_kernel<<<16384, 256, 0, stream>>>(h, lng, lnb, h16);

  // 3) projections
  gemm_nt<<<dim3(128, 8), 256, 0, stream>>>(h16, wq16, bq, q16, nullptr, nullptr, nullptr, 0);
  gemm_nt<<<dim3(4, 8),   256, 0, stream>>>(mem16, wk16, bk, k16, nullptr, nullptr, nullptr, 0);
  gemm_nt<<<dim3(4, 8),   256, 0, stream>>>(mem16, wv16, bv, v16, nullptr, nullptr, nullptr, 0);

  // 4) transpose V per head for attention B-matrix
  vtrans<<<2048, 256, 0, stream>>>(v16, vT16);

  // 5) attention (overwrites h16 region with o16)
  attn_kernel<<<4096, 128, 0, stream>>>(q16, k16, vT16, o16);

  // 6) output projection + gate + residual -> d_out (f32)
  gemm_nt<<<dim3(128, 8), 256, 0, stream>>>(o16, wo16, bo, nullptr, out, h, gate, 1);
}
